// BahdanauMonotonicAttention_51582557225638
// MI455X (gfx1250) — matrix operand order repaired and checked by reading, not compiled, not run on hardware
//
#include <hip/hip_runtime.h>
#include <hip/hip_bf16.h>

#define NB 32
#define TT 2048
#define EE 512
#define DD 1024
#define UU 256

typedef __attribute__((ext_vector_type(16))) __bf16 v16bf;
typedef __attribute__((ext_vector_type(8)))  float  v8f;

union V16BF { v16bf v; uint4 q[2]; };

__device__ __forceinline__ float fast_tanh(float x) {
    float e = __expf(2.0f * x);
    return 1.0f - 2.0f / (e + 1.0f);
}

// ---------------- prep: W_k -> transposed bf16 hi/lo split ----------------
__global__ __launch_bounds__(256)
void prep_wk_kernel(const float* __restrict__ Wk,
                    __bf16* __restrict__ hi, __bf16* __restrict__ lo) {
    int idx = blockIdx.x * 256 + threadIdx.x;   // 0 .. 256*512-1
    int u = idx >> 9;
    int e = idx & 511;
    float w = Wk[e * UU + u];
    __bf16 h = (__bf16)w;
    float r = w - (float)h;
    hi[u * EE + e] = h;
    lo[u * EE + e] = (__bf16)r;
}

// ---------------- prep: proj_q = queries @ W_q (fp32, tiny) ----------------
__global__ __launch_bounds__(256)
void prep_projq_kernel(const float* __restrict__ q, const float* __restrict__ Wq,
                       float* __restrict__ pq) {
    int n = blockIdx.x, u = threadIdx.x;
    const float* qr = q + n * DD;
    float acc = 0.f;
    for (int d = 0; d < DD; ++d) acc = fmaf(qr[d], Wq[d * UU + u], acc);
    pq[n * UU + u] = acc;
}

// ---------------- fused score: memory @ W_k (bf16x3 WMMA) + tanh.v + sigmoid ----------------
#define KCH  32   // K chunk staged in LDS
#define BSTR 40   // padded LDS row stride in halfs (32 + 8) -> conflict-free b128 reads

__global__ __launch_bounds__(256)
void score_kernel(const float* __restrict__ mem, const __bf16* __restrict__ BHg,
                  const __bf16* __restrict__ BLg, const float* __restrict__ pq,
                  const float* __restrict__ vvec, float* __restrict__ p_out) {
    __shared__ __bf16 shH[UU * BSTR];
    __shared__ __bf16 shL[UU * BSTR];

    int bx   = blockIdx.x;
    int n    = bx >> 4;        // 16 T-tiles of 128 per batch row
    int tb   = bx & 15;
    int tid  = threadIdx.x;
    int wave = tid >> 5;
    int lane = tid & 31;
    int lhalf = lane & 15;
    bool hi16 = lane >= 16;
    int trow = tb * 128 + wave * 16;

    v8f acc[16];
    v8f zero = {};
#pragma unroll
    for (int i = 0; i < 16; ++i) acc[i] = zero;

    // preload per-lane proj_q and v for each U tile (N = lane&15 within tile)
    float pqv[16], vv[16];
#pragma unroll
    for (int ut = 0; ut < 16; ++ut) {
        int u = ut * 16 + lhalf;
        pqv[ut] = pq[n * UU + u];
        vv[ut]  = vvec[u];
    }

    // A: lane holds row trow + (lane&15); K base offset 8 for upper half-wave
    const float* arow = mem + (size_t)(n * TT + trow + lhalf) * EE + (hi16 ? 8 : 0);

    for (int kc = 0; kc < EE; kc += KCH) {
        __syncthreads();
        {   // stage B chunk (all 256 u rows, KCH halfs each) into LDS
            int u = tid;
            const uint4* sh = (const uint4*)(BHg + u * EE + kc);
            const uint4* sl = (const uint4*)(BLg + u * EE + kc);
            uint4* dH = (uint4*)&shH[u * BSTR];
            uint4* dL = (uint4*)&shL[u * BSTR];
#pragma unroll
            for (int j = 0; j < 4; ++j) { dH[j] = sh[j]; dL[j] = sl[j]; }
        }
        __syncthreads();

        // A fragment: fp32 load + hi/lo bf16 split in registers
        V16BF ah, al;
        {
            const float* ap = arow + kc;
#pragma unroll
            for (int c = 0; c < 2; ++c) {        // chunks at K +0..7 and +16..23
                float4 f0 = ((const float4*)(ap + c * 16))[0];
                float4 f1 = ((const float4*)(ap + c * 16))[1];
                float xs[8] = {f0.x, f0.y, f0.z, f0.w, f1.x, f1.y, f1.z, f1.w};
#pragma unroll
                for (int j = 0; j < 8; ++j) {
                    float x = xs[j];
                    __bf16 h = (__bf16)x;
                    ah.v[c * 8 + j] = h;
                    al.v[c * 8 + j] = (__bf16)(x - (float)h);
                }
            }
        }

#pragma unroll
        for (int ut = 0; ut < 16; ++ut) {
            int u = ut * 16 + lhalf;
            // B: lane<16 holds K 0..7,16..23 of column u; lanes>=16 hold K 8..15,24..31
            const __bf16* bh = &shH[u * BSTR + (hi16 ? 8 : 0)];
            const __bf16* bl = &shL[u * BSTR + (hi16 ? 8 : 0)];
            V16BF Bh, Bl;
            Bh.q[0] = *(const uint4*)(bh);
            Bh.q[1] = *(const uint4*)(bh + 16);
            Bl.q[0] = *(const uint4*)(bl);
            Bl.q[1] = *(const uint4*)(bl + 16);
            acc[ut] = __builtin_amdgcn_wmma_f32_16x16x32_bf16(false, ah.v, false, Bh.v,
                                                              (short)0, acc[ut], false, false);
            acc[ut] = __builtin_amdgcn_wmma_f32_16x16x32_bf16(false, al.v, false, Bh.v,
                                                              (short)0, acc[ut], false, false);
            acc[ut] = __builtin_amdgcn_wmma_f32_16x16x32_bf16(false, ah.v, false, Bl.v,
                                                              (short)0, acc[ut], false, false);
        }
    }

    // epilogue: weights[t] = sum_u tanh(proj + proj_q) * v[u]; p = sigmoid(w)
    float wsum[8];
#pragma unroll
    for (int g = 0; g < 8; ++g) wsum[g] = 0.f;
#pragma unroll
    for (int ut = 0; ut < 16; ++ut) {
#pragma unroll
        for (int g = 0; g < 8; ++g) {
            float x = acc[ut][g] + pqv[ut];
            wsum[g] += fast_tanh(x) * vv[ut];
        }
    }
#pragma unroll
    for (int g = 0; g < 8; ++g) {
        float s = wsum[g];
        s += __shfl_xor(s, 1, 32);
        s += __shfl_xor(s, 2, 32);
        s += __shfl_xor(s, 4, 32);
        s += __shfl_xor(s, 8, 32);
        if (lhalf == 0) {                     // C/D layout: VGPR g -> row g (low half) / 8+g (high half)
            int t = trow + g + (hi16 ? 8 : 0);
            p_out[n * TT + t] = 1.0f / (1.0f + __expf(-s));
        }
    }
}

// ---------------- monotonic recursion: log-space scans, in-place p -> alignments ----------------
__global__ __launch_bounds__(256)
void scan_kernel(const float* __restrict__ prev, float* __restrict__ alig) {
    __shared__ float sc[256];
    int n = blockIdx.x, tid = threadIdx.x;
    float* row = alig + n * TT;
    const float* pr = prev + n * TT;
    int base = tid * 8;

    float p[8], incl[8];
#pragma unroll
    for (int i = 0; i < 8; ++i) p[i] = row[base + i];

    // scan 1: inclusive cumsum of log(clip(1-p, 1e-20, 1))
    float run = 0.f;
#pragma unroll
    for (int i = 0; i < 8; ++i) {
        float x = 1.0f - p[i];
        x = fminf(fmaxf(x, 1e-20f), 1.0f);
        run += logf(x);
        incl[i] = run;
    }
    sc[tid] = run; __syncthreads();
    for (int off = 1; off < 256; off <<= 1) {
        float tv = (tid >= off) ? sc[tid - off] : 0.f;
        __syncthreads();
        sc[tid] += tv;
        __syncthreads();
    }
    float exclT = sc[tid] - run;
    __syncthreads();

    float cp[8], qv[8];
#pragma unroll
    for (int i = 0; i < 8; ++i) {
        float e = exclT + (i > 0 ? incl[i - 1] : 0.f);   // exclusive logsum
        float c = expf(e);
        cp[i] = c;
        float cc = fminf(fmaxf(c, 1e-10f), 1.0f);
        qv[i] = pr[base + i] / cc;
    }

    // scan 2: inclusive cumsum of prev / clip(cumprod)
    run = 0.f;
#pragma unroll
    for (int i = 0; i < 8; ++i) { run += qv[i]; incl[i] = run; }
    sc[tid] = run; __syncthreads();
    for (int off = 1; off < 256; off <<= 1) {
        float tv = (tid >= off) ? sc[tid - off] : 0.f;
        __syncthreads();
        sc[tid] += tv;
        __syncthreads();
    }
    exclT = sc[tid] - run;
#pragma unroll
    for (int i = 0; i < 8; ++i)
        row[base + i] = p[i] * cp[i] * (exclT + incl[i]);
}

// ---------------- contexts = alignments @ memory (L2-resident second pass) ----------------
__global__ __launch_bounds__(1024)
void context_kernel(const float* __restrict__ mem, const float* __restrict__ alig,
                    float* __restrict__ ctx) {
    __shared__ float red[4][256];
    int n = blockIdx.x, ec = blockIdx.y;
    int tid = threadIdx.x;
    int el = tid & 255, tg = tid >> 8;
    int e = ec * 256 + el;
    const float* arow = alig + n * TT;
    const float* mcol = mem + (size_t)n * TT * EE + e;
    float acc = 0.f;
    int t0 = tg * 512;
    for (int t = t0; t < t0 + 512; ++t)
        acc = fmaf(arow[t], mcol[(size_t)t * EE], acc);
    red[tg][el] = acc; __syncthreads();
    if (tg == 0)
        ctx[n * EE + e] = red[0][el] + red[1][el] + red[2][el] + red[3][el];
}

extern "C" void kernel_launch(void* const* d_in, const int* in_sizes, int n_in,
                              void* d_out, int out_size, void* d_ws, size_t ws_size,
                              hipStream_t stream) {
    const float* queries = (const float*)d_in[0];   // [32,1,1024]
    const float* prev    = (const float*)d_in[1];   // [32,2048]
    const float* memory  = (const float*)d_in[2];   // [32,2048,512]
    const float* Wq      = (const float*)d_in[3];   // [1024,256]
    const float* Wk      = (const float*)d_in[4];   // [512,256]
    const float* v       = (const float*)d_in[5];   // [256]

    float* ctx  = (float*)d_out;            // contexts [32*512]
    float* alig = (float*)d_out + NB * EE;  // alignments [32*2048] (p_choose staged here)

    char* ws = (char*)d_ws;
    __bf16* BH = (__bf16*)ws;                         // 256 KB
    __bf16* BL = (__bf16*)(ws + 256 * 1024);          // 256 KB
    float*  pq = (float*)(ws + 512 * 1024);           // 32 KB

    prep_wk_kernel<<<(UU * EE) / 256, 256, 0, stream>>>(Wk, BH, BL);
    prep_projq_kernel<<<NB, UU, 0, stream>>>(queries, Wq, pq);
    score_kernel<<<NB * 16, 256, 0, stream>>>(memory, BH, BL, pq, v, alig);
    scan_kernel<<<NB, 256, 0, stream>>>(prev, alig);
    context_kernel<<<dim3(NB, 2), 1024, 0, stream>>>(memory, alig, ctx);
}
